// NeighborAttention_22170621182100
// MI455X (gfx1250) — compile-verified, matrix operand order checked by reading, not run on hardware
//
#include <hip/hip_runtime.h>
#include <hip/hip_bf16.h>
#include <cstdint>

// ---------------------------------------------------------------------------
// Types for CDNA5 WMMA (wave32, 16x16x32 bf16 -> f32)
// ---------------------------------------------------------------------------
typedef __attribute__((ext_vector_type(16))) __bf16 v16bf;
typedef __attribute__((ext_vector_type(8)))  __bf16 v8bf;
typedef __attribute__((ext_vector_type(8)))  float  v8f;

// ---------------------------------------------------------------------------
// bf16 helpers (bit-exact RNE conversion; no reliance on __bf16 arithmetic)
// ---------------------------------------------------------------------------
__device__ __forceinline__ __bf16 f2bf(float f) {
    unsigned u = __float_as_uint(f);
    unsigned r = u + 0x7FFFu + ((u >> 16) & 1u);   // round to nearest even
    unsigned short h = (unsigned short)(r >> 16);
    __bf16 b; __builtin_memcpy(&b, &h, 2); return b;
}
__device__ __forceinline__ float bf2f(__bf16 b) {
    unsigned short s; __builtin_memcpy(&s, &b, 2);
    return __uint_as_float(((unsigned)s) << 16);
}
__device__ __forceinline__ float gelu_erf(float x) {
    return 0.5f * x * (1.0f + erff(x * 0.70710678118654752f));
}
__device__ __forceinline__ void atomicMaxFloat(float* addr, float v) {
    if (v >= 0.0f) atomicMax((int*)addr, __float_as_int(v));
    else           atomicMin((unsigned int*)addr, __float_as_uint(v));
}

// ---------------------------------------------------------------------------
// WMMA fragment load.
// ISA 7.12.2, 16-bit A 16x32 (and symmetric B when B is stored as W[n][k]):
//   lane r = lane&15 selects the row (M for A, N for B),
//   g = lane>>4 selects the 8-wide K chunk: VGPR0-3 hold k in [8g,8g+8),
//   VGPR4-7 hold k in [16+8g, 24+8g).  Two 16-byte loads per lane.
// ---------------------------------------------------------------------------
__device__ __forceinline__ v16bf load_frag(const __bf16* base, int stride, int lane) {
    const int r = lane & 15, g = lane >> 4;
    const __bf16* p = base + (size_t)r * stride + 8 * g;
    v8bf lo = *reinterpret_cast<const v8bf*>(p);
    v8bf hi = *reinterpret_cast<const v8bf*>(p + 16);
    v16bf o;
#pragma unroll
    for (int i = 0; i < 8; ++i) { o[i] = lo[i]; o[i + 8] = hi[i]; }
    return o;
}

// D = A(16xK) * B^T(16 rows of K) with f32 accumulate.  A row-major with
// leading dim lda, B = weight rows W[n][k] with leading dim ldb.
__device__ __forceinline__ v8f wmma_rowcol(const __bf16* A, int lda,
                                           const __bf16* B, int ldb,
                                           int K, int lane) {
    v8f acc = {0.f, 0.f, 0.f, 0.f, 0.f, 0.f, 0.f, 0.f};
    for (int k = 0; k < K; k += 32) {
        v16bf a = load_frag(A + k, lda, lane);
        v16bf b = load_frag(B + k, ldb, lane);
        acc = __builtin_amdgcn_wmma_f32_16x16x32_bf16(
            false, a, false, b, (short)0, acc, false, false);
    }
    return acc;
}

// ---------------------------------------------------------------------------
// Kernel 0a: f32 -> bf16 weight conversion
// ---------------------------------------------------------------------------
__global__ void na_cvt_bf16(const float* __restrict__ src, __bf16* __restrict__ dst, int n) {
    int i = blockIdx.x * blockDim.x + threadIdx.x;
    if (i < n) dst[i] = f2bf(src[i]);
}

// Kernel 0b: Wb3 (4x128 f32) -> zero-padded 16x128 bf16 (rows 4..15 = 0)
__global__ void na_cvt_pad16(const float* __restrict__ src, __bf16* __restrict__ dst,
                             int rows_src, int cols) {
    int i = blockIdx.x * blockDim.x + threadIdx.x;
    if (i >= 16 * cols) return;
    int row = i / cols, col = i - row * cols;
    dst[i] = f2bf(row < rows_src ? src[row * cols + col] : 0.0f);
}

// ---------------------------------------------------------------------------
// Kernel 1: init accumulators
// ---------------------------------------------------------------------------
__global__ void na_init(float* __restrict__ S, float* __restrict__ denom,
                        float* __restrict__ nmax, int nS, int nD) {
    int i = blockIdx.x * blockDim.x + threadIdx.x;
    if (i < nS) S[i] = 0.0f;
    if (i < nD) { denom[i] = 0.0f; nmax[i] = -INFINITY; }
}

// ---------------------------------------------------------------------------
// Kernel 2: per-edge MLPs (bias MLP + value MLP) — 64 edges per block,
// 8 wave32 waves, all GEMMs (including head logits) via
// v_wmma_f32_16x16x32_bf16.  Per-wave output column tile nt == wave id is
// loop-invariant, so the compiler keeps the weight fragments in VGPRs.
// ---------------------------------------------------------------------------
__global__ __launch_bounds__(256)
void na_edge_mlp(const float* __restrict__ hV, const float* __restrict__ hE,
                 const int* __restrict__ center,
                 const __bf16* __restrict__ Wb1, const float* __restrict__ bb1,
                 const __bf16* __restrict__ Wb2, const float* __restrict__ bb2,
                 const __bf16* __restrict__ Wb3p, const float* __restrict__ bb3,
                 const __bf16* __restrict__ Wv1, const float* __restrict__ bv1,
                 const __bf16* __restrict__ Wv2, const float* __restrict__ bv2,
                 const __bf16* __restrict__ Wv3, const float* __restrict__ bv3,
                 float* __restrict__ wlog, __bf16* __restrict__ vbuf,
                 float* __restrict__ nmax, int E) {
    __shared__ __bf16 sX[64 * 384];   // [hV[center] | hE] tile, bf16
    __shared__ __bf16 sA[64 * 128];   // ping
    __shared__ __bf16 sB[64 * 128];   // pong
    __shared__ int    sCID[64];

    const int t = threadIdx.x, lane = t & 31, w = t >> 5;
    const int e0 = blockIdx.x * 64;

    if (t < 64) sCID[t] = (e0 + t < E) ? center[e0 + t] : 0;

    // prefetch next tile's edge features (global_prefetch_b8)
    if ((size_t)(e0 + 64) * 256 < (size_t)E * 256)
        __builtin_prefetch(hE + (size_t)(e0 + 64) * 256 + t * 64, 0, 1);
    __syncthreads();

    // ---- stage [hV[center] | hE] as bf16 ----
    for (int i = t; i < 64 * 384; i += 256) {
        int e = i / 384, c = i - e * 384;
        float v = 0.0f;
        if (e0 + e < E)
            v = (c < 128) ? hV[(size_t)sCID[e] * 128 + c]
                          : hE[(size_t)(e0 + e) * 256 + (c - 128)];
        sX[e * 384 + c] = f2bf(v);
    }
    __syncthreads();

    // ---- bias MLP stage 1: sA = gelu(sX(64x384) @ Wb1^T + bb1) ----
    {
        const int nt = w, n = nt * 16 + (lane & 15), g = lane >> 4;
        for (int mt = 0; mt < 4; ++mt) {
            v8f acc = wmma_rowcol(sX + mt * 16 * 384, 384, Wb1 + nt * 16 * 384, 384, 384, lane);
#pragma unroll
            for (int r = 0; r < 8; ++r) {
                int m = mt * 16 + r + 8 * g;
                sA[m * 128 + n] = f2bf(gelu_erf(acc[r] + bb1[n]));
            }
        }
    }
    __syncthreads();

    // ---- bias MLP stage 2: sB = gelu(sA @ Wb2^T + bb2) ----
    {
        const int nt = w, n = nt * 16 + (lane & 15), g = lane >> 4;
        for (int mt = 0; mt < 4; ++mt) {
            v8f acc = wmma_rowcol(sA + mt * 16 * 128, 128, Wb2 + nt * 16 * 128, 128, 128, lane);
#pragma unroll
            for (int r = 0; r < 8; ++r) {
                int m = mt * 16 + r + 8 * g;
                sB[m * 128 + n] = f2bf(gelu_erf(acc[r] + bb2[n]));
            }
        }
    }
    __syncthreads();

    // ---- per-edge per-head logit via WMMA: (sB @ Wb3p^T + bb3) / sqrt(32) ----
    // Wb3p is Wb3 zero-padded to 16x128; waves 0..3 each own one 16-edge tile.
    if (w < 4) {
        v8f acc = wmma_rowcol(sB + w * 16 * 128, 128, Wb3p, 128, 128, lane);
        int n = lane & 15, g = lane >> 4;
        if (n < 4) {
#pragma unroll
            for (int r = 0; r < 8; ++r) {
                int el = w * 16 + r + 8 * g;
                float lg = (acc[r] + bb3[n]) * 0.17677669529663687f;
                if (e0 + el < E) {
                    wlog[(size_t)(e0 + el) * 4 + n] = lg;
                    atomicMaxFloat(&nmax[(size_t)sCID[el] * 4 + n], lg);
                }
            }
        }
    }

    // ---- value MLP stage 1: sA = gelu(hE(64x256) @ Wv1^T + bv1) ----
    // (hE columns live at sX[:,128:384]; sA's previous readers synced above)
    {
        const int nt = w, n = nt * 16 + (lane & 15), g = lane >> 4;
        for (int mt = 0; mt < 4; ++mt) {
            v8f acc = wmma_rowcol(sX + mt * 16 * 384 + 128, 384, Wv1 + nt * 16 * 256, 256, 256, lane);
#pragma unroll
            for (int r = 0; r < 8; ++r) {
                int m = mt * 16 + r + 8 * g;
                sA[m * 128 + n] = f2bf(gelu_erf(acc[r] + bv1[n]));
            }
        }
    }
    __syncthreads();   // logits done reading sB, V1 written

    // ---- value MLP stage 2: sB = gelu(sA @ Wv2^T + bv2) ----
    {
        const int nt = w, n = nt * 16 + (lane & 15), g = lane >> 4;
        for (int mt = 0; mt < 4; ++mt) {
            v8f acc = wmma_rowcol(sA + mt * 16 * 128, 128, Wv2 + nt * 16 * 128, 128, 128, lane);
#pragma unroll
            for (int r = 0; r < 8; ++r) {
                int m = mt * 16 + r + 8 * g;
                sB[m * 128 + n] = f2bf(gelu_erf(acc[r] + bv2[n]));
            }
        }
    }
    __syncthreads();

    // ---- value MLP stage 3: vbuf = sB @ Wv3^T + bv3 (bf16 to global) ----
    {
        const int nt = w, n = nt * 16 + (lane & 15), g = lane >> 4;
        for (int mt = 0; mt < 4; ++mt) {
            v8f acc = wmma_rowcol(sB + mt * 16 * 128, 128, Wv3 + nt * 16 * 128, 128, 128, lane);
#pragma unroll
            for (int r = 0; r < 8; ++r) {
                int e = e0 + mt * 16 + r + 8 * g;
                if (e < E) vbuf[(size_t)e * 128 + n] = f2bf(acc[r] + bv3[n]);
            }
        }
    }
}

// ---------------------------------------------------------------------------
// Kernel 3: scatter — un-normalized softmax accumulation.
//   S[c]     += exp(w - max) * V      (per channel)
//   denom[c] += exp(w - max)          (per head)
// ---------------------------------------------------------------------------
__global__ __launch_bounds__(256)
void na_scatter(const float* __restrict__ wlog, const __bf16* __restrict__ vbuf,
                const int* __restrict__ center, const float* __restrict__ nmax,
                float* __restrict__ S, float* __restrict__ denom, int E) {
    long long idx = (long long)blockIdx.x * 256 + threadIdx.x;
    if (idx >= (long long)E * 128) return;
    int e = (int)(idx >> 7), d = (int)(idx & 127), h = d >> 5;
    int c = center[e];
    float ew = __expf(wlog[(size_t)e * 4 + h] - nmax[(size_t)c * 4 + h]);
    atomicAdd(&S[(size_t)c * 128 + d], ew * bf2f(vbuf[idx]));
    if ((d & 31) == 0) atomicAdd(&denom[(size_t)c * 4 + h], ew);
}

// ---------------------------------------------------------------------------
// Kernel 4: normalize + out = agg @ Wo^T   (16 nodes per block, WMMA)
// ---------------------------------------------------------------------------
__global__ __launch_bounds__(256)
void na_node_out(const float* __restrict__ S, const float* __restrict__ denom,
                 const __bf16* __restrict__ Wo, float* __restrict__ out, int N) {
    __shared__ __bf16 sAgg[16 * 128];
    const int t = threadIdx.x, lane = t & 31, w = t >> 5;
    const int n0 = blockIdx.x * 16;
    for (int i = t; i < 16 * 128; i += 256) {
        int nl = i >> 7, d = i & 127, h = d >> 5;
        int node = n0 + nl;
        float v = 0.0f;
        if (node < N) {
            float den = denom[(size_t)node * 4 + h];
            v = (den > 0.0f) ? S[(size_t)node * 128 + d] / den : 0.0f;
        }
        sAgg[i] = f2bf(v);
    }
    __syncthreads();
    // 8 waves: one 16x16 output tile each (N columns 16w .. 16w+15)
    v8f acc = wmma_rowcol(sAgg, 128, Wo + w * 16 * 128, 128, 128, lane);
    int n = w * 16 + (lane & 15), g = lane >> 4;
#pragma unroll
    for (int r = 0; r < 8; ++r) {
        int node = n0 + r + 8 * g;
        if (node < N) out[(size_t)node * 128 + n] = acc[r];
    }
}

// ---------------------------------------------------------------------------
// Host launcher
// ---------------------------------------------------------------------------
extern "C" void kernel_launch(void* const* d_in, const int* in_sizes, int n_in,
                              void* d_out, int out_size, void* d_ws, size_t ws_size,
                              hipStream_t stream) {
    const float* hV  = (const float*)d_in[0];
    const float* hE  = (const float*)d_in[1];
    const float* Wv1 = (const float*)d_in[2];  const float* bv1 = (const float*)d_in[3];
    const float* Wv2 = (const float*)d_in[4];  const float* bv2 = (const float*)d_in[5];
    const float* Wv3 = (const float*)d_in[6];  const float* bv3 = (const float*)d_in[7];
    const float* Wb1 = (const float*)d_in[8];  const float* bb1 = (const float*)d_in[9];
    const float* Wb2 = (const float*)d_in[10]; const float* bb2 = (const float*)d_in[11];
    const float* Wb3 = (const float*)d_in[12]; const float* bb3 = (const float*)d_in[13];
    const float* Wo  = (const float*)d_in[14];
    const int* center = (const int*)d_in[15];

    const int N = in_sizes[0] / 128;   // 20000
    const int E = in_sizes[1] / 256;   // 600000

    // workspace carve-out (256B aligned)
    char* base = (char*)d_ws; size_t off = 0;
    auto carve = [&](size_t bytes) { void* p = base + off; off = (off + bytes + 255) & ~(size_t)255; return p; };
    __bf16* Wb1b = (__bf16*)carve((size_t)128 * 384 * 2);
    __bf16* Wb2b = (__bf16*)carve((size_t)128 * 128 * 2);
    __bf16* Wb3p = (__bf16*)carve((size_t)16  * 128 * 2);
    __bf16* Wv1b = (__bf16*)carve((size_t)128 * 256 * 2);
    __bf16* Wv2b = (__bf16*)carve((size_t)128 * 128 * 2);
    __bf16* Wv3b = (__bf16*)carve((size_t)128 * 128 * 2);
    __bf16* Wob  = (__bf16*)carve((size_t)128 * 128 * 2);
    float*  nmax  = (float*)carve((size_t)N * 4 * 4);
    float*  denom = (float*)carve((size_t)N * 4 * 4);
    float*  S     = (float*)carve((size_t)N * 128 * 4);
    float*  wlog  = (float*)carve((size_t)E * 4 * 4);
    __bf16* vbuf  = (__bf16*)carve((size_t)E * 128 * 2);

    // 1) weight conversion
    na_cvt_bf16<<<(128 * 384 + 255) / 256, 256, 0, stream>>>(Wb1, Wb1b, 128 * 384);
    na_cvt_bf16<<<(128 * 128 + 255) / 256, 256, 0, stream>>>(Wb2, Wb2b, 128 * 128);
    na_cvt_pad16<<<(16 * 128 + 255) / 256, 256, 0, stream>>>(Wb3, Wb3p, 4, 128);
    na_cvt_bf16<<<(128 * 256 + 255) / 256, 256, 0, stream>>>(Wv1, Wv1b, 128 * 256);
    na_cvt_bf16<<<(128 * 128 + 255) / 256, 256, 0, stream>>>(Wv2, Wv2b, 128 * 128);
    na_cvt_bf16<<<(128 * 128 + 255) / 256, 256, 0, stream>>>(Wv3, Wv3b, 128 * 128);
    na_cvt_bf16<<<(128 * 128 + 255) / 256, 256, 0, stream>>>(Wo,  Wob,  128 * 128);

    // 2) init accumulators
    na_init<<<(N * 128 + 255) / 256, 256, 0, stream>>>(S, denom, nmax, N * 128, N * 4);

    // 3) per-edge MLPs
    na_edge_mlp<<<(E + 63) / 64, 256, 0, stream>>>(
        hV, hE, center,
        Wb1b, bb1, Wb2b, bb2, Wb3p, bb3,
        Wv1b, bv1, Wv2b, bv2, Wv3b, bv3,
        wlog, vbuf, nmax, E);

    // 4) softmax scatter accumulation
    long long work = (long long)E * 128;
    na_scatter<<<(unsigned)((work + 255) / 256), 256, 0, stream>>>(
        wlog, vbuf, center, nmax, S, denom, E);

    // 5) normalize + output projection
    na_node_out<<<(N + 15) / 16, 256, 0, stream>>>(S, denom, Wob, (float*)d_out, N);
}
